// Criterion_61349312856119
// MI455X (gfx1250) — matrix operand order repaired and checked by reading, not compiled, not run on hardware
//
#include <hip/hip_runtime.h>

typedef __attribute__((ext_vector_type(16))) _Float16 v16h;
typedef __attribute__((ext_vector_type(8)))  float    v8f;

#define B_   4
#define H_   384
#define W_   768
#define HB_  48
#define WB_  96
#define L_   (HB_*WB_)   // 4608
#define R_   (B_*L_)     // 18432
#define ND_  48
#define NP_  8
#define NPERM_ 1680
#define MAXD_ 320.0f

// ws accumulator slots (floats)
#define WS_SDISP 0
#define WS_CNTD  1
#define WS_SEPE  2
#define WS_SLOGP 3
#define WS_NVP   4
#define WS_SPROP 5
#define WS_NPROP 6
#define WS_SERR  7
#define WS_NERR  8

__device__ __forceinline__ float smooth_l1(float x) {
    float a = fabsf(x);
    return a < 1.0f ? 0.5f * x * x : a - 0.5f;
}

// Branchless, array-safe decode of the p-th 4-permutation of {0..7}
// (itertools lexicographic order). Fully unrolled; no dynamic indexing,
// so nothing is demoted to scratch.
__device__ __forceinline__ void decode_perm(int p, int d[4]) {
    int q[4];
    q[0] = p / 210; int r = p % 210;
    q[1] = r / 30;  r %= 30;
    q[2] = r / 5;   q[3] = r % 5;
    unsigned used = 0;
#pragma unroll
    for (int k = 0; k < 4; ++k) {
        int s = q[k];
        int idx = 0;
#pragma unroll
        for (int i = 0; i < 8; ++i) {
            bool fre = ((used >> i) & 1u) == 0u;
            bool hit = fre && (s == 0);
            idx = hit ? i : idx;
            s -= fre ? 1 : 0;
        }
        d[k] = idx;
        used |= 1u << idx;
    }
}

// ---------------- Kernel 1: l_disp + epe pixel sums ----------------
__global__ void k_pixel(const float* __restrict__ disp,
                        const float* __restrict__ disp_pred,
                        const float* __restrict__ tgt,
                        const unsigned char* __restrict__ valid,
                        float* __restrict__ ws) {
    const int n = B_ * H_ * W_;
    float sd = 0.f, se = 0.f, c = 0.f;
    for (int i = blockIdx.x * blockDim.x + threadIdx.x; i < n; i += gridDim.x * blockDim.x) {
        float g = valid[i] ? tgt[i] : 0.f;
        float m = (g > 0.f && g < MAXD_) ? 1.f : 0.f;
        sd += smooth_l1(disp_pred[i] * 4.f - g) * m;
        se += fabsf(disp[i] - g) * m;
        c  += m;
    }
    __shared__ float r0, r1, r2;
    if (threadIdx.x == 0) { r0 = 0.f; r1 = 0.f; r2 = 0.f; }
    __syncthreads();
    atomicAdd(&r0, sd); atomicAdd(&r1, se); atomicAdd(&r2, c);
    __syncthreads();
    if (threadIdx.x == 0) {
        atomicAdd(ws + WS_SDISP, r0);
        atomicAdd(ws + WS_SEPE,  r1);
        atomicAdd(ws + WS_CNTD,  r2);
    }
}

// ---------------- Kernel 2: l_init (soft-histogram cross entropy) ----------------
__global__ void k_init(const float* __restrict__ prob,
                       const float* __restrict__ tgt,
                       const unsigned char* __restrict__ valid,
                       const unsigned char* __restrict__ occ,
                       const unsigned char* __restrict__ occ2,
                       float* __restrict__ ws) {
    __shared__ float hist[64 * ND_];
    const int r = blockIdx.x * 64 + threadIdx.x;
    float* h = hist + threadIdx.x * ND_;
    for (int k = 0; k < ND_; ++k) h[k] = 0.f;

    const int b  = r / L_;
    const int rem = r % L_;
    const int hb = rem / WB_;
    const int wb = rem % WB_;
    int vcnt = 0;
    for (int e = 0; e < 64; ++e) {
        int m = e >> 3, nn = e & 7;
        int y = hb * 8 + m, x = wb * 8 + nn;
        int pix = (b * H_ + y) * W_ + x;
        float g = valid[pix] ? tgt[pix] : 0.f;
        g = fmaxf(g, 0.f);
        bool v = (g > 0.f) && (g < MAXD_) && (occ[pix] == 0);
        float coord = (float)x - g;
        v = v && (coord >= 0.f);
        int ci = (int)floorf(coord); if (ci < 0) ci = 0;
        v = v && (occ2[(b * H_ + y) * W_ + ci] == 0);
        float w = v ? 1.f : 0.f;
        vcnt += v ? 1 : 0;
        float t = g * 0.125f;
        int lo = (int)floorf(t);
        float hp = t - (float)lo;
        int hi = lo + 1; if (hi > ND_ - 1) hi = ND_ - 1;
        if (lo > ND_ - 1) lo = ND_ - 1;
        h[lo] += (1.f - hp) * w;
        h[hi] += hp * w;
    }
    float s = 0.f;
    for (int k = 0; k < ND_; ++k) s += h[k];
    float inv = 1.f / fmaxf(s, 0.001f);
    const float* pr = prob + (size_t)r * ND_;
    float lp = 0.f;
    for (int k = 0; k < ND_; ++k) lp += logf(fmaxf(pr[k], 1e-6f)) * h[k];
    lp = -lp * inv;
    atomicAdd(ws + WS_SLOGP, lp);
    atomicAdd(ws + WS_NVP, vcnt > 0 ? 1.f : 0.f);
}

// ---------------- Kernel 3: l_prop / l_err via WMMA permutation matching ----------------
__global__ void __launch_bounds__(256)
k_prop(const float* __restrict__ proposal,
       const float* __restrict__ tgt_disp,
       const unsigned char* __restrict__ valid,
       float* __restrict__ ws) {
    // 32 KB region: s_gt (phases 1-2) overlaid with s_rc/s_rp (phases 3-4)
    __shared__ float    s_u[8192];
    __shared__ _Float16 s_pm[128 * 32];   // 8 KB
    __shared__ float    s_pr[128 * 8];    // 4 KB
    __shared__ float    s_tg[128 * 4];    // 2 KB
    __shared__ unsigned s_sel[NPERM_];    // 6.6 KB one-hot selection words
    float* s_gt = s_u;                    // [128][64]
    float* s_rc = s_u;                    // [8][32][8]
    int*   s_rp = (int*)(s_u + 2048);     // [8][32][8]

    const int row0 = blockIdx.x * 128;
    const int tid  = threadIdx.x;

    // Phase 1: stage gt blocks, scaled proposals, and permutation selwords
    for (int q = tid; q < 128 * 64; q += 256) {
        int rl = q >> 6, e = q & 63;
        int row = row0 + rl;
        int b = row / L_, rem = row % L_;
        int hb = rem / WB_, wb = rem % WB_;
        int m = e >> 3, nn = e & 7;
        int pix = (b * H_ + hb * 8 + m) * W_ + wb * 8 + nn;
        __builtin_prefetch(tgt_disp + pix + 8, 0, 0);   // global_prefetch_b8
        s_gt[q] = valid[pix] ? tgt_disp[pix] : 0.f;
    }
    for (int q = tid; q < 128 * 8; q += 256)
        s_pr[q] = proposal[(size_t)row0 * 8 + q] * 8.f;
    for (int p = tid; p < NPERM_; p += 256) {
        int d[4]; decode_perm(p, d);
        s_sel[p] = (1u << d[0]) | (1u << (8 + d[1])) |
                   (1u << (16 + d[2])) | (1u << (24 + d[3]));
    }
    __syncthreads();

    // Phase 2: per-row order statistics, target prep, pairm (f16)
    if (tid < 128) {
        const float* gv = s_gt + tid * 64;
        float g4[4] = {0.f, 0.f, 0.f, 0.f};
        for (int i = 0; i < 64; ++i) {
            float v = gv[i];
            int rank = 0;
            for (int j = 0; j < 64; ++j) {
                float u = gv[j];
                rank += ((u > v) || (u == v && j < i)) ? 1 : 0;
            }
            if      (rank == 0)  g4[0] = v;
            else if (rank == 16) g4[1] = v;
            else if (rank == 32) g4[2] = v;
            else if (rank == 48) g4[3] = v;
        }
        float t[4];
        for (int k = 0; k < 4; ++k) t[k] = (g4[k] >= MAXD_) ? 0.f : g4[k];
        const float* pr = s_pr + tid * 8;
        float dist[4];
        for (int k = 0; k < 4; ++k) {
            if (t[k] == 0.f) dist[k] = 1e6f;
            else {
                float d = 3.4e38f;
                for (int j = 0; j < NP_; ++j) d = fminf(d, fabsf(t[k] - pr[j]));
                dist[k] = d;
            }
        }
        float ts[4];
        for (int k = 0; k < 4; ++k) {
            int pos = 0;
            for (int j = 0; j < 4; ++j)
                pos += ((dist[j] < dist[k]) || (dist[j] == dist[k] && j < k)) ? 1 : 0;
            ts[pos] = t[k];
        }
        for (int i = 0; i < 3; ++i) {
            float ti = ts[i];
            for (int j = i + 1; j < 4; ++j)
                if (ti > 0.f && fabsf(ts[j] - ti) < 8.f) ts[j] = 0.f;
        }
        for (int k = 0; k < 4; ++k) s_tg[tid * 4 + k] = ts[k];
        for (int k = 0; k < 4; ++k) {
            bool vt = ts[k] > 0.f;
            for (int j = 0; j < NP_; ++j) {
                float pm = vt ? fabsf(ts[k] - pr[j]) : 0.f;
                s_pm[tid * 32 + k * 8 + j] = (_Float16)pm;
            }
        }
    }
    __syncthreads();

    // Phase 3: cost = pairm @ selection via V_WMMA_F32_16X16X32_F16
    const int wave = tid >> 5;
    const int lane = tid & 31;
    const int Mrow = lane & 15;
    const int half = lane >> 4;

    v16h a;
    {
        const _Float16* pm = s_pm + (wave * 16 + Mrow) * 32;
        const int base = half * 8;
#pragma unroll
        for (int e = 0; e < 16; ++e) {
            int K = (e < 8) ? (base + e) : (16 + base + (e - 8)); // ISA 16-bit A 16x32 layout
            a[e] = pm[K];
        }
    }

    float bc[8]; int bp[8];
#pragma unroll
    for (int g = 0; g < 8; ++g) { bc[g] = 3.4e38f; bp[g] = 0; }

#pragma unroll 3
    for (int tile = 0; tile < 105; ++tile) {
        const int p = tile * 16 + Mrow;     // N column == perm index within tile
        const unsigned sel = s_sel[p];
        v16h bf;
#pragma unroll
        for (int e = 0; e < 16; ++e) {
            int K = (e < 8) ? (half * 8 + e) : (16 + half * 8 + (e - 8)); // B: same K order as A (ISA 16-bit table layout)
            bf[e] = ((sel >> K) & 1u) ? (_Float16)1.0f : (_Float16)0.0f;
        }
        v8f c = {};
        c = __builtin_amdgcn_wmma_f32_16x16x32_f16(false, a, false, bf,
                                                   (short)0, c, false, false);
#pragma unroll
        for (int g = 0; g < 8; ++g)
            if (c[g] < bc[g]) { bc[g] = c[g]; bp[g] = p; }
    }
#pragma unroll
    for (int g = 0; g < 8; ++g) {
        s_rc[(wave * 32 + lane) * 8 + g] = bc[g];
        s_rp[(wave * 32 + lane) * 8 + g] = bp[g];
    }
    __syncthreads();

    // Phase 4: reduce over N lanes per row (argmin, first-occurrence ties), final sums
    if (lane < 16) {
        const int M = lane;
        const int srcBase = (M < 8) ? 0 : 16;
        const int g = M & 7;
        float bestC = 3.4e38f; int bestP = 0x7fffffff;
        for (int n = 0; n < 16; ++n) {
            float cst = s_rc[(wave * 32 + srcBase + n) * 8 + g];
            int   p   = s_rp[(wave * 32 + srcBase + n) * 8 + g];
            if (cst < bestC || (cst == bestC && p < bestP)) { bestC = cst; bestP = p; }
        }
        const int rl = wave * 16 + M;
        int d[4]; decode_perm(bestP, d);
        const float* pr = s_pr + rl * 8;
        const float* ts = s_tg + rl * 4;
        float sp = 0.f, np = 0.f, se = 0.f, ne = 0.f;
#pragma unroll
        for (int k = 0; k < 4; ++k) {
            float t = ts[k];
            float src = pr[d[k]];
            if (t > 0.f && t < MAXD_) { sp += smooth_l1(src - t); np += 1.f; }
            bool vt = t > 0.f;
            float e = vt ? fabsf(src - t) : 1e5f;
            if (e != 1e5f) { se += e; ne += 1.f; }
        }
        atomicAdd(ws + WS_SPROP, sp);
        atomicAdd(ws + WS_NPROP, np);
        atomicAdd(ws + WS_SERR,  se);
        atomicAdd(ws + WS_NERR,  ne);
    }
}

// ---------------- Kernel 4: finalize 5 scalars ----------------
__global__ void k_final(const float* __restrict__ ws, float* __restrict__ out) {
    if (threadIdx.x == 0 && blockIdx.x == 0) {
        out[0] = ws[WS_SPROP] / (ws[WS_NPROP] + 1e-6f);
        out[1] = ws[WS_SERR]  / (ws[WS_NERR]  + 1e-6f);
        out[2] = ws[WS_SLOGP] / (ws[WS_NVP]   + 1e-6f);
        out[3] = ws[WS_SDISP] / fmaxf(ws[WS_CNTD], 1.f);
        out[4] = ws[WS_SEPE]  / fmaxf(ws[WS_CNTD], 1.f);
    }
}

extern "C" void kernel_launch(void* const* d_in, const int* in_sizes, int n_in,
                              void* d_out, int out_size, void* d_ws, size_t ws_size,
                              hipStream_t stream) {
    (void)in_sizes; (void)n_in; (void)out_size; (void)ws_size;
    const float* prob      = (const float*)d_in[0];
    const float* proposal  = (const float*)d_in[1];
    const float* disp      = (const float*)d_in[2];
    const float* disp_pred = (const float*)d_in[3];
    const float* tgt_disp  = (const float*)d_in[4];
    const unsigned char* valid = (const unsigned char*)d_in[5];
    const unsigned char* occ   = (const unsigned char*)d_in[6];
    const unsigned char* occ2  = (const unsigned char*)d_in[7];
    float* ws  = (float*)d_ws;
    float* out = (float*)d_out;

    hipMemsetAsync(ws, 0, 16 * sizeof(float), stream);
    k_pixel<<<1152, 256, 0, stream>>>(disp, disp_pred, tgt_disp, valid, ws);
    k_init <<<R_ / 64, 64, 0, stream>>>(prob, tgt_disp, valid, occ, occ2, ws);
    k_prop <<<R_ / 128, 256, 0, stream>>>(proposal, tgt_disp, valid, ws);
    k_final<<<1, 32, 0, stream>>>(ws, out);
}
